// HAN_1425929142444
// MI455X (gfx1250) — compile-verified, matrix operand order checked by reading, not run on hardware
//
#include <hip/hip_runtime.h>
#include <hip/hip_bf16.h>

typedef __attribute__((ext_vector_type(16))) _Float16 v16h;
typedef __attribute__((ext_vector_type(8)))  float    v8f;

#define Gc 3
#define Bc 4096
#define Nc 32
#define Dc 256
#define Cc 128
#define Hc 8
#define Ac 128
#define Lc 16
#define HCc 1024

// ---- CDNA5 async global->LDS helpers (ASYNCcnt-tracked, bypass VGPRs) -------
__device__ __forceinline__ void async_ld_b128(void* lds_dst, const void* gsrc) {
  unsigned      l = (unsigned)(size_t)lds_dst;       // low 32 bits = LDS offset
  unsigned long long g = (unsigned long long)(size_t)gsrc;
  asm volatile("global_load_async_to_lds_b128 %0, %1, off"
               :: "v"(l), "v"(g) : "memory");
}
__device__ __forceinline__ void wait_async0() {
#if __has_builtin(__builtin_amdgcn_s_wait_asynccnt)
  __builtin_amdgcn_s_wait_asynccnt(0);
#else
  asm volatile("s_wait_asynccnt 0x0" ::: "memory");
#endif
}

// ---------------- K0: fp32 -> f16 weight conversion, TRANSPOSED --------------
// kf16t[c][d]  (128 x 256), semf16t[a][d] (128 x 1024): column of B contiguous
__global__ __launch_bounds__(256) void kc_convert(const float* __restrict__ kw,
                                                  const float* __restrict__ semk,
                                                  _Float16* __restrict__ kf16t,
                                                  _Float16* __restrict__ semf16t) {
  int i = blockIdx.x * 256 + threadIdx.x;
  if (i < Dc * Cc) {
    int d = i >> 7, c = i & 127;
    kf16t[c * Dc + d] = (_Float16)kw[i];
  }
  if (i < HCc * Ac) {
    int d = i >> 7, a = i & 127;
    semf16t[a * HCc + d] = (_Float16)semk[i];
  }
}

// ---------------- K0b: M_node[d][h] = sum_c kernel[d][c] * ak_node[h][c] -----
__global__ __launch_bounds__(256) void km_node(const float* __restrict__ kw,
                                               const float* __restrict__ ak,
                                               float* __restrict__ Mn) {
  int d = threadIdx.x;  // one block of 256 threads
  for (int h = 0; h < Hc; ++h) {
    float acc = 0.f;
    for (int c = 0; c < Cc; ++c) acc += kw[d * Cc + c] * ak[h * (2 * Cc) + c];
    Mn[d * Hc + h] = acc;
  }
}

// ---------------- K1: node_scores[b][h] = targets[b,:] . M_node[:,h] ---------
__global__ __launch_bounds__(256) void k1_nodescore(const float* __restrict__ tg,
                                                    const float* __restrict__ Mn,
                                                    float* __restrict__ ns) {
  int gid = blockIdx.x * 256 + threadIdx.x;   // 0 .. B*H-1
  int b = gid >> 3, h = gid & 7;
  float acc = 0.f;
  for (int d = 0; d < Dc; ++d) acc += tg[b * Dc + d] * Mn[d * Hc + h];
  ns[gid] = acc;
}

// ---------------- K2: fused neigh-GEMM + attention + z -----------------------
// one block per (g,b); WMMA M=32 N=128 K=256
__global__ __launch_bounds__(256) void k2_neigh_attn(
    const float* __restrict__ contexts, const _Float16* __restrict__ kf16t,
    const float* __restrict__ attn_kernel, const float* __restrict__ attn_bias,
    const float* __restrict__ node_scores, float* __restrict__ inp) {
  __shared__ _Float16 sA[Nc * 32];    // 32x32 A k-step tile, [n][kk]  (2 KB)
  __shared__ _Float16 sBt[Cc * 32];   // B k-step tile, [c][kk]        (8 KB)
  __shared__ float    sN[Nc * Cc];    // neigh tile 32x128             (16 KB)
  __shared__ float    sS[Hc * Nc];    // scores
  __shared__ float    sC[Hc * Nc];    // softmax coefficients

  const int bid = blockIdx.x;
  const int g = bid / Bc;
  const int b = bid % Bc;
  const int t = threadIdx.x;
  const int lane = t & 31;
  const int wave = t >> 5;            // n-tile index 0..7
  const int lh = lane & 15;
  const int hi = lane >> 4;
  const int kb1 = hi * 8, kb2 = 16 + hi * 8;
  const int col = wave * 16 + lh;

  const float* Abase = contexts + (size_t)(g * Bc + b) * (Nc * Dc);

  v8f acc0 = {};
  v8f acc1 = {};

  for (int ks = 0; ks < 8; ++ks) {
    // B tile: async DMA straight into LDS (already f16, pre-transposed)
#pragma unroll
    for (int i = 0; i < 2; ++i) {
      int chunk = t * 2 + i;             // 0..511 b128 chunks
      int c = chunk >> 2, q = chunk & 3; // c row, q quarter-of-row
      async_ld_b128(&sBt[c * 32 + q * 8], &kf16t[c * Dc + ks * 32 + q * 8]);
    }
    // A tile: load f32, convert, store f16
#pragma unroll
    for (int i = 0; i < 4; ++i) {          // 32x32 floats -> f16
      int idx = t * 4 + i;
      int n = idx >> 5, kk = idx & 31;
      sA[n * 32 + kk] = (_Float16)Abase[n * Dc + ks * 32 + kk];
    }
    if (ks < 7) __builtin_prefetch(&Abase[(t & 31) * Dc + (ks + 1) * 32], 0, 1);
    wait_async0();
    __syncthreads();

    v16h a0, a1, bf;
#pragma unroll
    for (int e = 0; e < 8; ++e) {
      a0[e]     = sA[lh * 32 + kb1 + e];
      a0[e + 8] = sA[lh * 32 + kb2 + e];
      a1[e]     = sA[(16 + lh) * 32 + kb1 + e];
      a1[e + 8] = sA[(16 + lh) * 32 + kb2 + e];
      bf[e]     = sBt[col * 32 + kb1 + e];
      bf[e + 8] = sBt[col * 32 + kb2 + e];
    }
    acc0 = __builtin_amdgcn_wmma_f32_16x16x32_f16(false, a0, false, bf,
                                                  (short)0, acc0, false, false);
    acc1 = __builtin_amdgcn_wmma_f32_16x16x32_f16(false, a1, false, bf,
                                                  (short)0, acc1, false, false);
    __syncthreads();
  }

  // spill neigh tile (D layout: vgpr r -> row r+8*hi, col lane&15)
#pragma unroll
  for (int r = 0; r < 8; ++r) {
    int row = r + 8 * hi;
    sN[row * Cc + col]        = acc0[r];
    sN[(16 + row) * Cc + col] = acc1[r];
  }
  __syncthreads();

  // scores[h][n] = node_score + neigh . ak_neigh, leaky-relu 0.2
  {
    int n = t >> 3, h = t & 7;
    const float* akn = attn_kernel + h * (2 * Cc) + Cc;
    float dot = 0.f;
    for (int c = 0; c < Cc; ++c) dot += sN[n * Cc + c] * akn[c];
    float sc = node_scores[b * Hc + h] + dot;
    sS[h * Nc + n] = sc > 0.f ? sc : 0.2f * sc;
  }
  __syncthreads();
  // softmax over n per head
  {
    int n = t >> 3, h = t & 7;
    float mx = -1e30f;
    for (int j = 0; j < Nc; ++j) mx = fmaxf(mx, sS[h * Nc + j]);
    float sum = 0.f;
    for (int j = 0; j < Nc; ++j) sum += __expf(sS[h * Nc + j] - mx);
    sC[h * Nc + n] = __expf(sS[h * Nc + n] - mx) / sum;
  }
  __syncthreads();
  // z[h][c] = sum_n coef * neigh ; write inp[b][g][:]
  float* out = inp + (size_t)(b * Gc + g) * HCc;
#pragma unroll
  for (int i = 0; i < 4; ++i) {
    int d = t + 256 * i;
    int h = d >> 7, c = d & 127;
    float z = 0.f;
    for (int n = 0; n < Nc; ++n) z += sC[h * Nc + n] * sN[n * Cc + c];
    out[d] = z + attn_bias[d];
  }
}

// ---------------- K3: s[r] = tanh(inp[r,:] @ semK) . sem_att  (WMMA) ---------
__global__ __launch_bounds__(256) void k3_sem(const float* __restrict__ inp,
                                              const _Float16* __restrict__ semf16t,
                                              const float* __restrict__ sem_att,
                                              float* __restrict__ s_out) {
  __shared__ _Float16 sA[16 * 32];    // 1 KB, [r][kk]
  __shared__ _Float16 sBt[Ac * 32];   // 8 KB, [a][kk]
  __shared__ float    sT[16 * Ac];    // 8 KB tanh staging
  const int t = threadIdx.x, lane = t & 31, wave = t >> 5;
  const int lh = lane & 15, hi = lane >> 4;
  const int kb1 = hi * 8, kb2 = 16 + hi * 8;
  const int col = wave * 16 + lh;
  const size_t r0 = (size_t)blockIdx.x * 16;

  v8f acc = {};
  for (int ks = 0; ks < 32; ++ks) {
    // B tile async: semf16t rows [a][ks*32 .. ks*32+31]
#pragma unroll
    for (int i = 0; i < 2; ++i) {
      int chunk = t * 2 + i;
      int a = chunk >> 2, q = chunk & 3;
      async_ld_b128(&sBt[a * 32 + q * 8], &semf16t[a * HCc + ks * 32 + q * 8]);
    }
#pragma unroll
    for (int i = 0; i < 2; ++i) {
      int idx = t * 2 + i;          // 512 elems
      int r = idx >> 5, kk = idx & 31;
      sA[r * 32 + kk] = (_Float16)inp[(r0 + r) * HCc + ks * 32 + kk];
    }
    wait_async0();
    __syncthreads();
    v16h a, bf;
#pragma unroll
    for (int e = 0; e < 8; ++e) {
      a[e]      = sA[lh * 32 + kb1 + e];
      a[e + 8]  = sA[lh * 32 + kb2 + e];
      bf[e]     = sBt[col * 32 + kb1 + e];
      bf[e + 8] = sBt[col * 32 + kb2 + e];
    }
    acc = __builtin_amdgcn_wmma_f32_16x16x32_f16(false, a, false, bf,
                                                 (short)0, acc, false, false);
    __syncthreads();
  }
#pragma unroll
  for (int r = 0; r < 8; ++r) {
    int row = r + 8 * hi;
    sT[row * Ac + col] = tanhf(acc[r]);
  }
  __syncthreads();
  if (t < 16) {
    float sum = 0.f;
    for (int a2 = 0; a2 < Ac; ++a2) sum += sT[t * Ac + a2] * sem_att[a2];
    s_out[r0 + t] = sum;
  }
}

// ---------------- K4: softmax over g, pooled, final dense --------------------
__global__ __launch_bounds__(256) void k4_pool(const float* __restrict__ inp,
                                               const float* __restrict__ s,
                                               const float* __restrict__ dense_w,
                                               const float* __restrict__ dense_b,
                                               float* __restrict__ out) {
  __shared__ float sP[HCc];
  __shared__ float sRed[256];
  const int b = blockIdx.x, t = threadIdx.x;
  float s0 = s[b * 3 + 0], s1 = s[b * 3 + 1], s2 = s[b * 3 + 2];
  float mx = fmaxf(s0, fmaxf(s1, s2));
  float e0 = __expf(s0 - mx), e1 = __expf(s1 - mx), e2 = __expf(s2 - mx);
  float inv = 1.f / (e0 + e1 + e2);
  float a0 = e0 * inv, a1 = e1 * inv, a2 = e2 * inv;
  const float* r = inp + (size_t)b * 3 * HCc;
#pragma unroll
  for (int i = 0; i < 4; ++i) {
    int d = t + 256 * i;
    sP[d] = a0 * r[d] + a1 * r[HCc + d] + a2 * r[2 * HCc + d];
  }
  __syncthreads();
  int o = t & 15, seg = t >> 4;
  float acc = 0.f;
  for (int j = 0; j < 64; ++j) {
    int d = seg * 64 + j;
    acc += sP[d] * dense_w[d * Lc + o];
  }
  sRed[seg * 16 + o] = acc;
  __syncthreads();
  if (t < 16) {
    float tot = dense_b[t];
    for (int j = 0; j < 16; ++j) tot += sRed[j * 16 + t];
    out[b * Lc + t] = tot;
  }
}

extern "C" void kernel_launch(void* const* d_in, const int* in_sizes, int n_in,
                              void* d_out, int out_size, void* d_ws, size_t ws_size,
                              hipStream_t stream) {
  (void)in_sizes; (void)n_in; (void)out_size; (void)ws_size;
  const float* targets     = (const float*)d_in[0];
  const float* contexts    = (const float*)d_in[1];
  const float* kernelW     = (const float*)d_in[2];
  const float* attn_kernel = (const float*)d_in[3];
  const float* attn_bias   = (const float*)d_in[4];
  const float* sem_kernel  = (const float*)d_in[5];
  const float* sem_att     = (const float*)d_in[6];
  const float* dense_w     = (const float*)d_in[7];
  const float* dense_b     = (const float*)d_in[8];
  float* out = (float*)d_out;

  char* ws = (char*)d_ws;
  _Float16* kf16t   = (_Float16*)(ws + 0);        // 64 KB  (transposed)
  _Float16* semf16t = (_Float16*)(ws + 65536);    // 256 KB (transposed)
  float*    Mn      = (float*)(ws + 327680);      // 8 KB
  float*    ns      = (float*)(ws + 335872);      // 128 KB
  float*    s_ws    = (float*)(ws + 466944);      // 48 KB
  float*    inp     = (float*)(ws + 524288);      // 48 MB

  kc_convert<<<512, 256, 0, stream>>>(kernelW, sem_kernel, kf16t, semf16t);
  km_node<<<1, 256, 0, stream>>>(kernelW, attn_kernel, Mn);
  k1_nodescore<<<(Bc * Hc) / 256, 256, 0, stream>>>(targets, Mn, ns);
  k2_neigh_attn<<<Gc * Bc, 256, 0, stream>>>(contexts, kf16t, attn_kernel,
                                             attn_bias, ns, inp);
  k3_sem<<<(Bc * Gc) / 16, 256, 0, stream>>>(inp, semf16t, sem_att, s_ws);
  k4_pool<<<Bc, 256, 0, stream>>>(inp, s_ws, dense_w, dense_b, out);
}
